// CantorExpert_74569222193920
// MI455X (gfx1250) — compile-verified, matrix operand order checked by reading, not run on hardware
//
#include <hip/hip_runtime.h>
#include <math.h>

typedef __attribute__((ext_vector_type(16))) __bf16 v16bf;
typedef __attribute__((ext_vector_type(8)))  float  v8f;
typedef __attribute__((ext_vector_type(4)))  int    vint4;

#define NB      16
#define PTOK    4096
#define DTOK    1024
#define N_SEL   2048
#define EDIM    256
#define SL      64
#define HID     128
#define QBLK    128
#define ROW0    1024
#define SLICE0  192

// ---- workspace layout (bytes) ----
#define OFF_QWB    0u                       // 256*64  bf16 = 32768
#define OFF_KWB    32768u                   // 32768
#define OFF_VWB    65536u                   // 32768
#define OFF_MQWB   98304u                   // 256*256 bf16 = 131072
#define OFF_KA     229376u                  // 5*256 f32 = 5120
#define OFF_W2     234496u                  // 5120
#define OFF_WGEO   239616u                  // 1024
#define OFF_MEAN   240640u                  // 16*256 f32 = 16384
#define OFF_BSCALE 257024u                  // 64
#define OFF_QB     262144u                  // 16*2048*256 bf16 = 16777216
#define OFF_KB     (OFF_QB  + 16777216u)
#define OFF_VTB    (OFF_KB  + 16777216u)
#define OFF_KGEO   (OFF_VTB + 16777216u)    // 16*2048 f32 = 131072
#define OFF_VGEO   (OFF_KGEO + 131072u)
#define OFF_OPRE   (OFF_VGEO + 131072u)     // 16*2048*256 f32 = 33554432

#define AS1 __attribute__((address_space(1)))
#define AS3 __attribute__((address_space(3)))

// ---- helpers ----
__device__ __forceinline__ __bf16 f2bf(float f) {
    unsigned u = __float_as_uint(f);
    unsigned r = (u + 0x7FFFu + ((u >> 16) & 1u)) >> 16;
    unsigned short s = (unsigned short)r;
    return *reinterpret_cast<__bf16*>(&s);
}
__device__ __forceinline__ float bf2f(__bf16 b) {
    unsigned short s = *reinterpret_cast<unsigned short*>(&b);
    unsigned u = ((unsigned)s) << 16;
    return __uint_as_float(u);
}

// Async global->LDS copies (CDNA5 GLOBAL_LOAD_ASYNC_TO_LDS_*, tracked by ASYNCcnt).
// Builtin signatures (from hipcc diagnostics): (T AS1*, T AS3*, Imm offset, Imm cpol),
// T = int4 for b128, int for b32.
__device__ __forceinline__ void async_ld_b128(void* lds, const void* gptr) {
#if __has_builtin(__builtin_amdgcn_global_load_async_to_lds_b128)
    __builtin_amdgcn_global_load_async_to_lds_b128(
        (AS1 vint4*)gptr, (AS3 vint4*)lds, 0, 0);
#else
    *reinterpret_cast<uint4*>(lds) = *reinterpret_cast<const uint4*>(gptr);
#endif
}
__device__ __forceinline__ void async_ld_b32(void* lds, const void* gptr) {
#if __has_builtin(__builtin_amdgcn_global_load_async_to_lds_b32)
    __builtin_amdgcn_global_load_async_to_lds_b32(
        (AS1 int*)gptr, (AS3 int*)lds, 0, 0);
#else
    *reinterpret_cast<unsigned*>(lds) = *reinterpret_cast<const unsigned*>(gptr);
#endif
}
__device__ __forceinline__ void wait_async0() {
#if __has_builtin(__builtin_amdgcn_s_wait_asynccnt)
    __builtin_amdgcn_s_wait_asynccnt(0);
#else
    asm volatile("s_wait_asynccnt 0" ::: "memory");
#endif
}

// ISA 7.12.2 16-bit A/B fragment: lane L holds row (L&15); halfs 0..7 = K[k0..k0+7],
// halfs 8..15 = K[k0+16..k0+23], where k0 = kofs + ((L>>4)<<3). Two 16B loads.
__device__ __forceinline__ v16bf load_frag(const __bf16* mat, int stride, int row, int kofs, int lane) {
    const __bf16* p = mat + (size_t)(row + (lane & 15)) * stride + kofs + ((lane >> 4) << 3);
    union { v16bf v; uint4 q[2]; } u;
    u.q[0] = *reinterpret_cast<const uint4*>(p);
    u.q[1] = *reinterpret_cast<const uint4*>(p + 16);
    return u.v;
}
__device__ __forceinline__ v8f wmma_bf16(v16bf a, v16bf b, v8f c) {
    return __builtin_amdgcn_wmma_f32_16x16x32_bf16(false, a, false, b, (short)0, c, false, false);
}

// =====================================================================
__global__ void zero_kernel(float* out, int n) {
    int i = blockIdx.x * 256 + threadIdx.x;
    if (i < n) out[i] = 0.0f;
}

// =====================================================================
__global__ __launch_bounds__(256)
void prep_kernel(const float* __restrict__ penta,
                 const float* __restrict__ mha_kw, const float* __restrict__ mha_kb,
                 const float* __restrict__ mha_vw, const float* __restrict__ mha_vb,
                 const float* __restrict__ mha_ow,
                 const float* __restrict__ q_w, const float* __restrict__ k_w,
                 const float* __restrict__ v_w, const float* __restrict__ mha_qw,
                 char* __restrict__ ws)
{
    __shared__ float sva[5 * EDIM];
    __shared__ float snorm[5];
    const int tid = threadIdx.x;
    const float role[5] = {1.0f, -0.75f, 0.75f, 0.75f, -0.75f};

    if (tid < 5) {
        float s = 0.f;
        for (int d = 0; d < EDIM; ++d) { float v = penta[tid * EDIM + d]; s += v * v; }
        snorm[tid] = fmaxf(sqrtf(s), 1e-12f);
    }
    __syncthreads();

    float* wgeo = (float*)(ws + OFF_WGEO);
    {   // w_geo[d] = sum_v role[v] * penta[v][d] / ||penta[v]||
        float s = 0.f;
        for (int v = 0; v < 5; ++v) s += role[v] * penta[v * EDIM + tid] / snorm[v];
        wgeo[tid] = s;
    }

    float* ka = (float*)(ws + OFF_KA);
    for (int t = tid; t < 5 * EDIM; t += 256) {
        int v = t >> 8, o = t & 255;
        float sk = mha_kb[o], sv = mha_vb[o];
        for (int d = 0; d < EDIM; ++d) {
            float p = penta[v * EDIM + d];
            sk += p * mha_kw[o * EDIM + d];
            sv += p * mha_vw[o * EDIM + d];
        }
        ka[t] = sk;
        sva[t] = sv;
    }
    __syncthreads();

    float* W2 = (float*)(ws + OFF_W2);   // W2[v][o] = sum_d va[v][d]*mha_ow[o][d]
    for (int t = tid; t < 5 * EDIM; t += 256) {
        int v = t >> 8, o = t & 255;
        float s = 0.f;
        for (int d = 0; d < EDIM; ++d) s += sva[v * EDIM + d] * mha_ow[o * EDIM + d];
        W2[t] = s;
    }

    __bf16* qwb  = (__bf16*)(ws + OFF_QWB);
    __bf16* kwb  = (__bf16*)(ws + OFF_KWB);
    __bf16* vwb  = (__bf16*)(ws + OFF_VWB);
    __bf16* mqwb = (__bf16*)(ws + OFF_MQWB);
    for (int i = tid; i < EDIM * SL; i += 256) {
        qwb[i] = f2bf(q_w[i]);
        kwb[i] = f2bf(k_w[i]);
        vwb[i] = f2bf(v_w[i]);
    }
    for (int i = tid; i < EDIM * EDIM; i += 256) mqwb[i] = f2bf(mha_qw[i]);
}

// =====================================================================
// Per block: one batch, 128 rows. WMMA for Q0/K/V (K=64) and qa (K=256).
#define QKV_SMEM (16384u + 131072u + 65536u + 65536u + 2560u)
__global__ __launch_bounds__(256)
void qkv_kernel(const float* __restrict__ tokens,
                const float* __restrict__ q_b, const float* __restrict__ k_b,
                const float* __restrict__ v_b, const float* __restrict__ mha_qb,
                const float* __restrict__ mha_ob, char* __restrict__ ws)
{
    extern __shared__ char smem[];
    __bf16* sx   = (__bf16*)(smem);             // 128 x 64
    float*  sQ0  = (float*) (smem + 16384);     // 128 x 256 f32
    __bf16* sQ0h = (__bf16*)(smem + 147456);    // 128 x 256 bf16
    __bf16* sqa  = (__bf16*)(smem + 212992);    // 128 x 256 bf16
    float*  satt = (float*) (smem + 278528);    // 128 x 5

    const int b = blockIdx.y;
    const int qbase = blockIdx.x * QBLK;
    const int tid = threadIdx.x;
    const int lane = tid & 31;
    const int qrow0 = (tid >> 5) * 16;
    const int n = lane & 15;
    const int mb = (lane >> 4) << 3;

    const __bf16* qwb  = (const __bf16*)(ws + OFF_QWB);
    const __bf16* kwb  = (const __bf16*)(ws + OFF_KWB);
    const __bf16* vwb  = (const __bf16*)(ws + OFF_VWB);
    const __bf16* mqwb = (const __bf16*)(ws + OFF_MQWB);
    const float*  ka   = (const float*)(ws + OFF_KA);
    const float*  W2   = (const float*)(ws + OFF_W2);
    const float*  wgeo = (const float*)(ws + OFF_WGEO);
    __bf16* Qb  = (__bf16*)(ws + OFF_QB);
    __bf16* Kb  = (__bf16*)(ws + OFF_KB);
    __bf16* Vtb = (__bf16*)(ws + OFF_VTB);
    float* kgeo = (float*)(ws + OFF_KGEO);
    float* vgeo = (float*)(ws + OFF_VGEO);

    // load token slice -> bf16 LDS
    for (int i = tid; i < QBLK * SL; i += 256) {
        int r = i >> 6, c = i & 63;
        float v = tokens[((size_t)b * PTOK + ROW0 + qbase + r) * DTOK + SLICE0 + c];
        sx[r * SL + c] = f2bf(v);
    }
    __syncthreads();

    // ---- Q0 = x @ q_w.T + q_b ----
    for (int nt = 0; nt < 16; ++nt) {
        v8f a = {};
        #pragma unroll
        for (int kc = 0; kc < 2; ++kc) {
            v16bf fa = load_frag(sx,  SL, qrow0,   kc * 32, lane);
            v16bf fb = load_frag(qwb, SL, nt * 16, kc * 32, lane);
            a = wmma_bf16(fa, fb, a);
        }
        int nn = nt * 16 + n;
        float bias = q_b[nn];
        #pragma unroll
        for (int r = 0; r < 8; ++r) {
            float v = a[r] + bias;
            sQ0 [(qrow0 + mb + r) * EDIM + nn] = v;
            sQ0h[(qrow0 + mb + r) * EDIM + nn] = f2bf(v);
        }
    }

    // ---- K = x @ k_w.T + k_b, k_geo ----
    {
        float nrm[8], dot[8];
        #pragma unroll
        for (int r = 0; r < 8; ++r) { nrm[r] = 0.f; dot[r] = 0.f; }
        for (int nt = 0; nt < 16; ++nt) {
            v8f a = {};
            #pragma unroll
            for (int kc = 0; kc < 2; ++kc) {
                v16bf fa = load_frag(sx,  SL, qrow0,   kc * 32, lane);
                v16bf fb = load_frag(kwb, SL, nt * 16, kc * 32, lane);
                a = wmma_bf16(fa, fb, a);
            }
            int nn = nt * 16 + n;
            float bias = k_b[nn], wg = wgeo[nn];
            #pragma unroll
            for (int r = 0; r < 8; ++r) {
                float v = a[r] + bias;
                Kb[((size_t)b * N_SEL + qbase + qrow0 + mb + r) * EDIM + nn] = f2bf(v);
                nrm[r] += v * v;
                dot[r] += v * wg;
            }
        }
        #pragma unroll
        for (int r = 0; r < 8; ++r) {
            float s1 = nrm[r], s2 = dot[r];
            #pragma unroll
            for (int m = 1; m <= 8; m <<= 1) {
                s1 += __shfl_xor(s1, m, 32);
                s2 += __shfl_xor(s2, m, 32);
            }
            if (n == 0)
                kgeo[(size_t)b * N_SEL + qbase + qrow0 + mb + r] = s2 / fmaxf(sqrtf(s1), 1e-12f);
        }
    }

    // ---- V = x @ v_w.T + v_b, stored transposed; v_geo ----
    {
        float nrm[8], dot[8];
        #pragma unroll
        for (int r = 0; r < 8; ++r) { nrm[r] = 0.f; dot[r] = 0.f; }
        for (int nt = 0; nt < 16; ++nt) {
            v8f a = {};
            #pragma unroll
            for (int kc = 0; kc < 2; ++kc) {
                v16bf fa = load_frag(sx,  SL, qrow0,   kc * 32, lane);
                v16bf fb = load_frag(vwb, SL, nt * 16, kc * 32, lane);
                a = wmma_bf16(fa, fb, a);
            }
            int nn = nt * 16 + n;
            float bias = v_b[nn], wg = wgeo[nn];
            #pragma unroll
            for (int r = 0; r < 8; ++r) {
                float v = a[r] + bias;
                Vtb[((size_t)b * EDIM + nn) * N_SEL + qbase + qrow0 + mb + r] = f2bf(v);
                nrm[r] += v * v;
                dot[r] += v * wg;
            }
        }
        #pragma unroll
        for (int r = 0; r < 8; ++r) {
            float s1 = nrm[r], s2 = dot[r];
            #pragma unroll
            for (int m = 1; m <= 8; m <<= 1) {
                s1 += __shfl_xor(s1, m, 32);
                s2 += __shfl_xor(s2, m, 32);
            }
            if (n == 0)
                vgeo[(size_t)b * N_SEL + qbase + qrow0 + mb + r] = s2 / fmaxf(sqrtf(s1), 1e-12f);
        }
    }
    __syncthreads();

    // ---- qa = Q0 @ mha_qw.T + mha_qb ----
    for (int nt = 0; nt < 16; ++nt) {
        v8f a = {};
        #pragma unroll
        for (int kc = 0; kc < 8; ++kc) {
            v16bf fa = load_frag(sQ0h, EDIM, qrow0,   kc * 32, lane);
            v16bf fb = load_frag(mqwb, EDIM, nt * 16, kc * 32, lane);
            a = wmma_bf16(fa, fb, a);
        }
        int nn = nt * 16 + n;
        float bias = mha_qb[nn];
        #pragma unroll
        for (int r = 0; r < 8; ++r)
            sqa[(qrow0 + mb + r) * EDIM + nn] = f2bf(a[r] + bias);
    }
    __syncthreads();

    // ---- attnv = softmax(qa @ ka.T / 16) over 5 ----
    if (tid < QBLK) {
        float lg[5];
        #pragma unroll
        for (int v = 0; v < 5; ++v) {
            float s = 0.f;
            for (int d = 0; d < EDIM; ++d) s += bf2f(sqa[tid * EDIM + d]) * ka[v * EDIM + d];
            lg[v] = s * 0.0625f;
        }
        float mx = lg[0];
        #pragma unroll
        for (int v = 1; v < 5; ++v) mx = fmaxf(mx, lg[v]);
        float sum = 0.f;
        #pragma unroll
        for (int v = 0; v < 5; ++v) { lg[v] = __expf(lg[v] - mx); sum += lg[v]; }
        float inv = 1.f / sum;
        #pragma unroll
        for (int v = 0; v < 5; ++v) satt[tid * 5 + v] = lg[v] * inv;
    }
    __syncthreads();

    // ---- Q = Q0 + attnv @ W2 + mha_ob  -> bf16 workspace ----
    for (int i = tid; i < QBLK * EDIM; i += 256) {
        int r = i >> 8, c = i & 255;
        float q = sQ0[i] + mha_ob[c];
        #pragma unroll
        for (int v = 0; v < 5; ++v) q += satt[r * 5 + v] * W2[v * EDIM + c];
        Qb[((size_t)b * N_SEL + qbase + r) * EDIM + c] = f2bf(q);
    }
}

// =====================================================================
// Flash attention: grid (16 q-tiles, 16 batches), 8 waves x 16 queries.
// Double-buffered async global->LDS staging of K / V^T / k_geo chunks.
#define ATTN_SMEM (65536u + 32768u + 32768u + 256u + 8192u)

__device__ __forceinline__ void stage_chunk(const __bf16* __restrict__ Kb,
                                            const __bf16* __restrict__ Vt,
                                            const float* __restrict__ kgeo,
                                            __bf16* dK, __bf16* dV, float* dKg,
                                            int kb0, int tid)
{
    #pragma unroll
    for (int i = tid; i < 1024; i += 256) {            // K: 32 x 256 bf16
        int r = i >> 5, c = (i & 31) << 3;
        async_ld_b128(dK + r * EDIM + c, Kb + (size_t)(kb0 + r) * EDIM + c);
    }
    #pragma unroll
    for (int i = tid; i < 1024; i += 256) {            // V^T: 256 x 32 bf16
        int col = i >> 2, c = (i & 3) << 3;
        async_ld_b128(dV + col * 32 + c, Vt + (size_t)col * N_SEL + kb0 + c);
    }
    if (tid < 32) async_ld_b32(dKg + tid, kgeo + kb0 + tid);
}

__global__ __launch_bounds__(256)
void attn_kernel(char* __restrict__ ws)
{
    extern __shared__ char smem[];
    __bf16* sQ  = (__bf16*)(smem);            // 128 x 256
    __bf16* sK  = (__bf16*)(smem + 65536);    // 2 x (32 x 256)
    __bf16* sVt = (__bf16*)(smem + 98304);    // 2 x (256 x 32)
    float*  sKg = (float*) (smem + 131072);   // 2 x 32
    __bf16* sP  = (__bf16*)(smem + 131328);   // 8 waves x 16 x 32

    const int b = blockIdx.y;
    const int qbase = blockIdx.x * QBLK;
    const int tid = threadIdx.x;
    const int wave = tid >> 5, lane = tid & 31;
    const int qrow0 = wave * 16;
    const int n = lane & 15;
    const int mb = (lane >> 4) << 3;

    const __bf16* Qb = (const __bf16*)(ws + OFF_QB)  + (size_t)b * N_SEL * EDIM;
    const __bf16* Kb = (const __bf16*)(ws + OFF_KB)  + (size_t)b * N_SEL * EDIM;
    const __bf16* Vt = (const __bf16*)(ws + OFF_VTB) + (size_t)b * EDIM * N_SEL;
    const float* kgeo = (const float*)(ws + OFF_KGEO) + b * N_SEL;
    const float* vgeo = (const float*)(ws + OFF_VGEO) + b * N_SEL;
    float* opre = (float*)(ws + OFF_OPRE) + (size_t)b * N_SEL * EDIM;

    // stage Q tile (128x256 bf16) asynchronously
    for (int i = tid; i < (QBLK * EDIM) / 8; i += 256) {
        int r = i >> 5, c = (i & 31) << 3;
        async_ld_b128(sQ + r * EDIM + c, Qb + (size_t)(qbase + r) * EDIM + c);
    }
    // stage first K/V^T chunk into buffer 0
    stage_chunk(Kb, Vt, kgeo, sK, sVt, sKg, 0, tid);

    v8f zero = {};
    v8f acc[16];
    #pragma unroll
    for (int t = 0; t < 16; ++t) acc[t] = zero;
    float mrow[8], lrow[8];
    #pragma unroll
    for (int r = 0; r < 8; ++r) { mrow[r] = -1e30f; lrow[r] = 0.f; }

    __bf16* sPw = sP + wave * 512;

    wait_async0();
    __syncthreads();

    for (int kt = 0; kt < N_SEL / 32; ++kt) {
        const int slot = kt & 1;
        if (kt + 1 < N_SEL / 32)   // overlap: async-stage the next chunk while computing
            stage_chunk(Kb, Vt, kgeo, sK + (slot ^ 1) * (32 * EDIM),
                        sVt + (slot ^ 1) * (EDIM * 32), sKg + (slot ^ 1) * 32,
                        (kt + 1) * 32, tid);

        const __bf16* cK  = sK  + slot * (32 * EDIM);
        const __bf16* cV  = sVt + slot * (EDIM * 32);
        const float*  cKg = sKg + slot * 32;

        // S tiles: 16 queries x 32 keys, K-dim 256
        v8f s0 = zero, s1 = zero;
        #pragma unroll
        for (int kc = 0; kc < 8; ++kc) {
            v16bf aq = load_frag(sQ, EDIM, qrow0, kc * 32, lane);
            v16bf b0 = load_frag(cK, EDIM, 0,     kc * 32, lane);
            v16bf b1 = load_frag(cK, EDIM, 16,    kc * 32, lane);
            s0 = wmma_bf16(aq, b0, s0);
            s1 = wmma_bf16(aq, b1, s1);
        }

        const float kg0 = cKg[n]      * 0.0625f;
        const float kg1 = cKg[16 + n] * 0.0625f;
        float corr[8];
        #pragma unroll
        for (int r = 0; r < 8; ++r) {
            float l0 = s0[r] * kg0, l1 = s1[r] * kg1;
            float rm = fmaxf(l0, l1);
            #pragma unroll
            for (int m = 1; m <= 8; m <<= 1) rm = fmaxf(rm, __shfl_xor(rm, m, 32));
            float mnew = fmaxf(mrow[r], rm);
            float c = __expf(mrow[r] - mnew);
            float p0 = __expf(l0 - mnew);
            float p1 = __expf(l1 - mnew);
            float rs = p0 + p1;
            #pragma unroll
            for (int m = 1; m <= 8; m <<= 1) rs += __shfl_xor(rs, m, 32);
            lrow[r] = lrow[r] * c + rs;
            mrow[r] = mnew;
            corr[r] = c;
            sPw[(mb + r) * 32 + n]      = f2bf(p0);
            sPw[(mb + r) * 32 + 16 + n] = f2bf(p1);
        }
        asm volatile("s_wait_dscnt 0" ::: "memory");

        // O += P(16x32) @ V(32x256)
        v16bf ap = load_frag(sPw, 32, 0, 0, lane);
        #pragma unroll
        for (int t = 0; t < 16; ++t) {
            v16bf bv = load_frag(cV, 32, t * 16, 0, lane);
            v8f c = acc[t];
            #pragma unroll
            for (int r = 0; r < 8; ++r) c[r] *= corr[r];
            acc[t] = wmma_bf16(ap, bv, c);
        }

        wait_async0();     // our next-chunk async copies have landed
        __syncthreads();   // everyone's have
    }

    // epilogue: O/l * v_geo[q]
    #pragma unroll
    for (int r = 0; r < 8; ++r) {
        int q = qbase + qrow0 + mb + r;
        float sc = vgeo[q] / lrow[r];
        #pragma unroll
        for (int t = 0; t < 16; ++t)
            opre[(size_t)q * EDIM + t * 16 + n] = acc[t][r] * sc;
    }
}

// =====================================================================
__global__ void mean_kernel(char* __restrict__ ws)
{
    int b = blockIdx.x, d = threadIdx.x; // 256 threads
    const float* opre = (const float*)(ws + OFF_OPRE) + (size_t)b * N_SEL * EDIM;
    float s = 0.f;
    for (int q = 0; q < N_SEL; ++q) s += opre[(size_t)q * EDIM + d];
    ((float*)(ws + OFF_MEAN))[b * EDIM + d] = s * (1.0f / N_SEL);
}

__global__ void gate_kernel(const float* __restrict__ g1_w, const float* __restrict__ g1_b,
                            const float* __restrict__ ln_g, const float* __restrict__ ln_b,
                            const float* __restrict__ g2_w, const float* __restrict__ g2_b,
                            const float* __restrict__ alpha, char* __restrict__ ws)
{
    __shared__ float red[HID];
    __shared__ float smu, svar;
    int b = blockIdx.x, j = threadIdx.x; // 128 threads
    const float* mean = (const float*)(ws + OFF_MEAN) + b * EDIM;
    float h = g1_b[j];
    for (int d = 0; d < EDIM; ++d) h += mean[d] * g1_w[j * EDIM + d];
    red[j] = h; __syncthreads();
    if (j == 0) { float s = 0.f; for (int i = 0; i < HID; ++i) s += red[i]; smu = s / HID; }
    __syncthreads();
    float hc = h - smu;
    red[j] = hc * hc; __syncthreads();
    if (j == 0) { float s = 0.f; for (int i = 0; i < HID; ++i) s += red[i]; svar = s / HID; }
    __syncthreads();
    float hn = hc * rsqrtf(svar + 1e-5f) * ln_g[j] + ln_b[j];
    float hg = 0.5f * hn * (1.f + erff(hn * 0.70710678118654752f));  // exact gelu
    red[j] = hg * g2_w[j]; __syncthreads();
    if (j == 0) {
        float s = 0.f; for (int i = 0; i < HID; ++i) s += red[i];
        float logit = s + g2_b[0];
        float ga = 1.f / (1.f + __expf(-logit));
        float sa = 1.f / (1.f + __expf(-alpha[0]));
        ((float*)(ws + OFF_BSCALE))[b] = sa * ga;
    }
}

// out_full[b, 1024+q, :] = (opre[b,q,:]*scale) @ out_w.T + out_b
__global__ void final_kernel(const float* __restrict__ out_w, const float* __restrict__ out_b,
                             const char* __restrict__ ws, float* __restrict__ out)
{
    __shared__ float srow[EDIM];
    int b = blockIdx.y, q = blockIdx.x, s = threadIdx.x; // 64 threads
    const float* opre = (const float*)(ws + OFF_OPRE) + ((size_t)b * N_SEL + q) * EDIM;
    for (int i = s; i < EDIM; i += 64) srow[i] = opre[i];
    __syncthreads();
    float sc = ((const float*)(ws + OFF_BSCALE))[b];
    float acc = 0.f;
    for (int d = 0; d < EDIM; ++d) acc += srow[d] * out_w[s * EDIM + d];
    out[(((size_t)b * PTOK) + ROW0 + q) * SL + s] = acc * sc + out_b[s];
}

__global__ void geo_kernel(const char* __restrict__ ws, float* __restrict__ out)
{
    int i = blockIdx.x * 256 + threadIdx.x; // 32768
    int b = i >> 11, q = i & 2047;
    out[(size_t)NB * PTOK * SL + (size_t)b * PTOK + ROW0 + q] =
        ((const float*)(ws + OFF_KGEO))[i];
}

// =====================================================================
extern "C" void kernel_launch(void* const* d_in, const int* in_sizes, int n_in,
                              void* d_out, int out_size, void* d_ws, size_t ws_size,
                              hipStream_t stream)
{
    (void)in_sizes; (void)n_in; (void)ws_size;
    const float* tokens = (const float*)d_in[0];
    const float* penta  = (const float*)d_in[2];
    const float* q_w    = (const float*)d_in[3];
    const float* q_b    = (const float*)d_in[4];
    const float* mha_qw = (const float*)d_in[5];
    const float* mha_qb = (const float*)d_in[6];
    const float* mha_kw = (const float*)d_in[7];
    const float* mha_kb = (const float*)d_in[8];
    const float* mha_vw = (const float*)d_in[9];
    const float* mha_vb = (const float*)d_in[10];
    const float* mha_ow = (const float*)d_in[11];
    const float* mha_ob = (const float*)d_in[12];
    const float* k_w    = (const float*)d_in[13];
    const float* k_b    = (const float*)d_in[14];
    const float* v_w    = (const float*)d_in[15];
    const float* v_b    = (const float*)d_in[16];
    const float* out_w  = (const float*)d_in[17];
    const float* out_b  = (const float*)d_in[18];
    const float* alpha  = (const float*)d_in[19];
    const float* g1_w   = (const float*)d_in[20];
    const float* g1_b   = (const float*)d_in[21];
    const float* ln_g   = (const float*)d_in[22];
    const float* ln_b   = (const float*)d_in[23];
    const float* g2_w   = (const float*)d_in[24];
    const float* g2_b   = (const float*)d_in[25];
    char* ws = (char*)d_ws;
    float* out = (float*)d_out;

    zero_kernel<<<(out_size + 255) / 256, 256, 0, stream>>>(out, out_size);
    prep_kernel<<<1, 256, 0, stream>>>(penta, mha_kw, mha_kb, mha_vw, mha_vb, mha_ow,
                                       q_w, k_w, v_w, mha_qw, ws);
    qkv_kernel<<<dim3(N_SEL / QBLK, NB), 256, QKV_SMEM, stream>>>(
        tokens, q_b, k_b, v_b, mha_qb, mha_ob, ws);
    attn_kernel<<<dim3(N_SEL / QBLK, NB), 256, ATTN_SMEM, stream>>>(ws);
    mean_kernel<<<NB, EDIM, 0, stream>>>(ws);
    gate_kernel<<<NB, HID, 0, stream>>>(g1_w, g1_b, ln_g, ln_b, g2_w, g2_b, alpha, ws);
    final_kernel<<<dim3(N_SEL, NB), SL, 0, stream>>>(out_w, out_b, ws, out);
    geo_kernel<<<(NB * N_SEL) / 256, 256, 0, stream>>>(ws, out);
}
